// ContrastiveLoss_44839458570876
// MI455X (gfx1250) — matrix operand order repaired and checked by reading, not compiled, not run on hardware
//
#include <hip/hip_runtime.h>
#include <math.h>

#define BATCH     4
#define CCH       32
#define HW        (512 * 512)
#define N_INST    100
#define N_PAD     112          // 7 * 16, padded instance count for WMMA tiling
#define STR       33           // LDS row stride (pad 32 -> 33 to avoid bank conflicts)
#define DELTA_VAR 0.75f
#define DELTA_DIST 2.0f

typedef __attribute__((ext_vector_type(16))) _Float16 v16h;
typedef __attribute__((ext_vector_type(8)))  float    v8f;

// -------- workspace layout (in floats) --------
#define WS_SUMS   0                                   // BATCH * N_PAD * CCH
#define WS_CNTS   (WS_SUMS + BATCH * N_PAD * CCH)     // BATCH * N_PAD
#define WS_MEANS  (WS_CNTS + BATCH * N_PAD)           // BATCH * N_PAD * CCH
#define WS_VAR    (WS_MEANS + BATCH * N_PAD * CCH)    // BATCH
#define WS_DIST   (WS_VAR + BATCH)                    // BATCH
#define WS_REG    (WS_DIST + BATCH)                   // BATCH
#define WS_TOTAL  (WS_REG + BATCH)

// ---------------------------------------------------------------------------
__global__ void dl_zero_ws(float* ws, int n) {
    int i = blockIdx.x * blockDim.x + threadIdx.x;
    if (i < n) ws[i] = 0.0f;
}

// ---------------------------------------------------------------------------
// Pass 1: per-instance counts & channel sums. 32 blocks per batch image,
// 256 threads, each thread handles 8 groups of 4 pixels (float4 streams).
// LDS scatter-accumulate (stride-33), then atomicAdd partials to global.
__global__ void dl_pass1_sums(const float* __restrict__ feat,
                              const int*   __restrict__ lbl,
                              float* __restrict__ g_sums,
                              float* __restrict__ g_cnts) {
    __shared__ float    s_sum[N_PAD * STR];
    __shared__ unsigned s_cnt[N_PAD];

    const int tid = threadIdx.x;
    for (int i = tid; i < N_PAD * STR; i += 256) s_sum[i] = 0.0f;
    for (int i = tid; i < N_PAD; i += 256) s_cnt[i] = 0u;
    __syncthreads();

    const int b   = blockIdx.x >> 5;          // / 32
    const int blk = blockIdx.x & 31;
    const int blockBase = blk * 8192;         // pixels per block
    const float* fb = feat + (size_t)b * CCH * HW;
    const int*   lb = lbl  + (size_t)b * HW;

    for (int it = 0; it < 8; ++it) {
        const int p = blockBase + (it * 256 + tid) * 4;
        const int4 l4 = *(const int4*)(lb + p);
        atomicAdd(&s_cnt[l4.x], 1u);
        atomicAdd(&s_cnt[l4.y], 1u);
        atomicAdd(&s_cnt[l4.z], 1u);
        atomicAdd(&s_cnt[l4.w], 1u);
#pragma unroll
        for (int c = 0; c < CCH; ++c) {
            const float4 f4 = *(const float4*)(fb + (size_t)c * HW + p);
            atomicAdd(&s_sum[l4.x * STR + c], f4.x);
            atomicAdd(&s_sum[l4.y * STR + c], f4.y);
            atomicAdd(&s_sum[l4.z * STR + c], f4.z);
            atomicAdd(&s_sum[l4.w * STR + c], f4.w);
        }
    }
    __syncthreads();

    float* gs = g_sums + (size_t)b * N_PAD * CCH;
    float* gc = g_cnts + (size_t)b * N_PAD;
    for (int idx = tid; idx < N_PAD * CCH; idx += 256) {
        const float v = s_sum[(idx >> 5) * STR + (idx & 31)];
        if (v != 0.0f) atomicAdd(&gs[idx], v);
    }
    for (int i = tid; i < N_PAD; i += 256)
        if (s_cnt[i]) atomicAdd(&gc[i], (float)s_cnt[i]);
}

// ---------------------------------------------------------------------------
__device__ inline float dl_block_reduce(float v, float* s_red, int tid) {
    s_red[tid] = v;
    __syncthreads();
    for (int off = 128; off > 0; off >>= 1) {
        if (tid < off) s_red[tid] += s_red[tid + off];
        __syncthreads();
    }
    const float r = s_red[0];
    __syncthreads();
    return r;
}

// ---------------------------------------------------------------------------
// Pass 2: means, Gram matrix via v_wmma_f32_16x16x32_f16, distance + reg terms.
// One 256-thread (8-wave) block per batch image. 7x7 = 49 tiles of 16x16,
// one WMMA per tile (K = 32 = channel count).
__global__ void dl_pass2_dist(const float* __restrict__ g_sums,
                              const float* __restrict__ g_cnts,
                              float* __restrict__ g_means,
                              float* __restrict__ ws) {
    __shared__ float    s_mean[N_PAD * STR];
    __shared__ _Float16 s_meanh[N_PAD * STR];
    __shared__ float    s_sq[N_PAD];
    __shared__ float    s_red[256];

    const int b   = blockIdx.x;
    const int tid = threadIdx.x;
    const float* gs = g_sums + (size_t)b * N_PAD * CCH;
    const float* gc = g_cnts + (size_t)b * N_PAD;
    float*       gm = g_means + (size_t)b * N_PAD * CCH;

    for (int idx = tid; idx < N_PAD * CCH; idx += 256) {
        const int i = idx >> 5, c = idx & 31;
        const float cnt = gc[i];
        const float m = (i < N_INST && cnt > 0.0f) ? gs[idx] / cnt : 0.0f;
        s_mean[i * STR + c]  = m;
        s_meanh[i * STR + c] = (_Float16)m;
        gm[idx] = m;
    }
    __syncthreads();

    if (tid < N_PAD) {
        float s = 0.0f;
#pragma unroll
        for (int c = 0; c < CCH; ++c) {
            const float v = s_mean[tid * STR + c];
            s += v * v;
        }
        s_sq[tid] = s;
    }
    __syncthreads();

    const int lane = tid & 31;
    const int wave = tid >> 5;        // 0..7, uniform per wave
    const int n    = lane & 15;
    const int hi   = lane >> 4;

    float acc = 0.0f;
    for (int t = wave; t < 49; t += 8) {   // uniform trip count within a wave
        const int i0 = (t / 7) * 16;
        const int j0 = (t % 7) * 16;

        v16h av, bv;
#pragma unroll
        for (int v = 0; v < 8; ++v) {
            // 16-bit A 16x32 layout: vgpr v<4 -> K = 2v + 8*hi ; v>=4 -> K = 16 + 2(v-4) + 8*hi
            const int ka = (v < 4) ? (hi * 8 + v * 2) : (16 + hi * 8 + (v - 4) * 2);
            av[2 * v]     = s_meanh[(i0 + n) * STR + ka];
            av[2 * v + 1] = s_meanh[(i0 + n) * STR + ka + 1];
            // B 32x16 layout: same K order as A: vgpr v<4 -> K = 2v + 8*hi ; v>=4 -> K = 16 + 2(v-4) + 8*hi
            const int kb = ka;
            bv[2 * v]     = s_meanh[(j0 + n) * STR + kb];     // B[k][n] = M[j0+n][k]
            bv[2 * v + 1] = s_meanh[(j0 + n) * STR + kb + 1];
        }
        v8f cacc = {};
        cacc = __builtin_amdgcn_wmma_f32_16x16x32_f16(
            /*neg_a=*/false, av, /*neg_b=*/false, bv,
            /*c_mod=*/(short)0, cacc, /*reuse_a=*/false, /*reuse_b=*/false);

#pragma unroll
        for (int r = 0; r < 8; ++r) {
            const int i = i0 + r + hi * 8;   // C layout: vgpr r -> M=r (lanes 0-15), M=r+8 (lanes 16-31)
            const int j = j0 + n;
            if (i < N_INST && j < N_INST && i != j) {
                const float d2 = s_sq[i] + s_sq[j] - 2.0f * cacc[r];
                const float d  = sqrtf(d2 > 0.0f ? d2 : 1.0f);
                const float h  = fmaxf(2.0f * DELTA_DIST - d, 0.0f);
                acc += h * h;
            }
        }
    }

    const float dsum = dl_block_reduce(acc, s_red, tid);
    if (tid == 0) ws[WS_DIST + b] = dsum;

    const float rsum = dl_block_reduce((tid < N_INST) ? sqrtf(s_sq[tid]) : 0.0f, s_red, tid);
    if (tid == 0) ws[WS_REG + b] = rsum;
}

// ---------------------------------------------------------------------------
// Pass 3: variance term. 128 blocks per batch, 256 threads, 2 groups of 4
// pixels per thread. Means table cached in LDS (stride-33).
__global__ void dl_pass3_var(const float* __restrict__ feat,
                             const int*   __restrict__ lbl,
                             const float* __restrict__ g_means,
                             const float* __restrict__ g_cnts,
                             float* __restrict__ g_var) {
    __shared__ float s_mean[N_INST * STR];
    __shared__ float s_icnt[N_INST];
    __shared__ float s_red[256];

    const int tid = threadIdx.x;
    const int b   = blockIdx.x >> 7;     // / 128
    const int blk = blockIdx.x & 127;

    const float* gm = g_means + (size_t)b * N_PAD * CCH;
    const float* gc = g_cnts  + (size_t)b * N_PAD;
    for (int idx = tid; idx < N_INST * CCH; idx += 256)
        s_mean[(idx >> 5) * STR + (idx & 31)] = gm[idx];
    for (int i = tid; i < N_INST; i += 256) {
        const float c = gc[i];
        s_icnt[i] = (c > 0.0f) ? 1.0f / c : 0.0f;
    }
    __syncthreads();

    const float* fb = feat + (size_t)b * CCH * HW;
    const int*   lb = lbl  + (size_t)b * HW;
    const int blockBase = blk * 2048;

    float acc = 0.0f;
    for (int it = 0; it < 2; ++it) {
        const int p = blockBase + (it * 256 + tid) * 4;
        const int4 l4 = *(const int4*)(lb + p);
        float s0 = 0.f, s1 = 0.f, s2 = 0.f, s3 = 0.f;
#pragma unroll
        for (int c = 0; c < CCH; ++c) {
            const float4 f4 = *(const float4*)(fb + (size_t)c * HW + p);
            const float d0 = f4.x - s_mean[l4.x * STR + c];
            const float d1 = f4.y - s_mean[l4.y * STR + c];
            const float d2 = f4.z - s_mean[l4.z * STR + c];
            const float d3 = f4.w - s_mean[l4.w * STR + c];
            s0 += d0 * d0; s1 += d1 * d1; s2 += d2 * d2; s3 += d3 * d3;
        }
        float h;
        h = fmaxf(sqrtf(s0) - DELTA_VAR, 0.0f); acc += h * h * s_icnt[l4.x];
        h = fmaxf(sqrtf(s1) - DELTA_VAR, 0.0f); acc += h * h * s_icnt[l4.y];
        h = fmaxf(sqrtf(s2) - DELTA_VAR, 0.0f); acc += h * h * s_icnt[l4.z];
        h = fmaxf(sqrtf(s3) - DELTA_VAR, 0.0f); acc += h * h * s_icnt[l4.w];
    }

    const float vsum = dl_block_reduce(acc, s_red, tid);
    if (tid == 0) atomicAdd(&g_var[b], vsum);
}

// ---------------------------------------------------------------------------
__global__ void dl_combine(const float* __restrict__ ws, float* __restrict__ out) {
    if (blockIdx.x == 0 && threadIdx.x == 0) {
        float acc = 0.0f;
        for (int b = 0; b < BATCH; ++b) {
            const float v = ws[WS_VAR  + b] / (float)N_INST;
            const float d = ws[WS_DIST + b] / (float)(N_INST * (N_INST - 1));
            const float r = ws[WS_REG  + b] / (float)N_INST;
            acc += 1.0f * v + 1.0f * d + 0.001f * r;   // ALPHA, BETA, GAMMA
        }
        out[0] = acc / (float)BATCH;
    }
}

// ---------------------------------------------------------------------------
extern "C" void kernel_launch(void* const* d_in, const int* in_sizes, int n_in,
                              void* d_out, int out_size, void* d_ws, size_t ws_size,
                              hipStream_t stream) {
    (void)in_sizes; (void)n_in; (void)out_size; (void)ws_size;
    const float* feat = (const float*)d_in[0];
    const int*   lbl  = (const int*)d_in[1];   // n_instances (d_in[2]) is fixed 100 in the reference
    float* ws  = (float*)d_ws;
    float* out = (float*)d_out;

    dl_zero_ws<<<(WS_TOTAL + 255) / 256, 256, 0, stream>>>(ws, WS_TOTAL);
    dl_pass1_sums<<<BATCH * 32, 256, 0, stream>>>(feat, lbl, ws + WS_SUMS, ws + WS_CNTS);
    dl_pass2_dist<<<BATCH, 256, 0, stream>>>(ws + WS_SUMS, ws + WS_CNTS, ws + WS_MEANS, ws);
    dl_pass3_var<<<BATCH * 128, 256, 0, stream>>>(feat, lbl, ws + WS_MEANS, ws + WS_CNTS, ws + WS_VAR);
    dl_combine<<<1, 32, 0, stream>>>(ws, out);
}
